// ContextWindowPredictor_28621662060658
// MI455X (gfx1250) — compile-verified, matrix operand order checked by reading, not run on hardware
//
#include <hip/hip_runtime.h>
#include <hip/hip_bf16.h>

// MI455X / gfx1250, wave32.
// GEMM1 on v_wmma_f32_16x16x32_f16 (f32 accum); A-panel gathered into LDS via
// global_load_async_to_lds_b128 (ASYNCcnt); fused bias+erf-GELU+W2 epilogue.
// Workspace: [0,4MiB) W1 transposed->f16, [4MiB,36MiB) hidden_states->f16.

typedef __attribute__((ext_vector_type(16))) _Float16 v16h;
typedef __attribute__((ext_vector_type(8)))  _Float16 v8h;
typedef __attribute__((ext_vector_type(4)))  _Float16 v4h;
typedef __attribute__((ext_vector_type(8)))  float    v8f;

union Frag { v16h v; struct { v8h lo, hi; } p; };

#define HD   1024      // H
#define KD   2048      // 2H (GEMM1 K)
#define PP   4096      // pairs per batch
#define BM   64        // rows (pairs) per block
#define ROWH 2056      // padded LDS row length in halves (2048 + 8 -> bank skew)

// ---- pre-kernel 1: W1 (2048,1024) f32 row-major -> W1T (1024,2048) f16 -----
__global__ __launch_bounds__(256) void w1_cvt_transpose(
    const float* __restrict__ W1, _Float16* __restrict__ W1T) {
  int idx = blockIdx.x * 256 + threadIdx.x;   // output index, n-major
  int n = idx >> 11;                          // /2048
  int k = idx & (KD - 1);
  W1T[idx] = (_Float16)W1[k * HD + n];
}

// ---- pre-kernel 2: hidden_states f32 -> f16 (vectorized x4) ----------------
__global__ __launch_bounds__(256) void hs_cvt(
    const float* __restrict__ hs, _Float16* __restrict__ o) {
  size_t i = ((size_t)blockIdx.x * 256 + threadIdx.x) * 4;
  float4 f = *(const float4*)(hs + i);
  v4h hv = { (_Float16)f.x, (_Float16)f.y, (_Float16)f.z, (_Float16)f.w };
  *(v4h*)(o + i) = hv;
}

// ---- fragment load helpers -------------------------------------------------
__device__ __forceinline__ void loadB(Frag& f, const _Float16* p) {
  f.p.lo = *(const v8h*)p;
  f.p.hi = *(const v8h*)(p + 16);
}

// ---- main kernel -----------------------------------------------------------
__global__ __launch_bounds__(256) void pair_mlp(
    const _Float16* __restrict__ hsF16, const long long* __restrict__ pairs,
    const _Float16* __restrict__ W1T, const float* __restrict__ b1,
    const float* __restrict__ W2, const float* __restrict__ b2,
    float* __restrict__ out)
{
  extern __shared__ char smem[];
  _Float16* Alds      = (_Float16*)smem;                 // BM*ROWH halves
  float*    logitsLds = (float*)(smem + BM * ROWH * 2);  // BM*2 floats

  const int tid    = threadIdx.x;
  const int gpBase = blockIdx.x * BM;
  const int wave   = tid >> 5;
  const int lane   = tid & 31;

  if (tid < BM * 2) logitsLds[tid] = 0.0f;

  // --- Gather the 64x2048 f16 A panel straight into LDS (async DMA path). ---
  // Each wave stages 8 rows; per row, two 2048B source rows (the pair) land
  // in the two halves of the padded LDS row. 16B per lane per op.
  {
    const char* hsb  = (const char*)hsF16;
    const unsigned ldsA = (unsigned)(uintptr_t)(void*)Alds;  // LDS byte offset
    #pragma unroll 1
    for (int i = 0; i < 8; ++i) {
      const int row = wave * 8 + i;
      const int gp  = gpBase + row;
      const int b   = gp >> 12;             // / 4096
      const int p   = gp & (PP - 1);
      const long long* pr = pairs + ((size_t)b * PP + p) * 2;  // wave-uniform
      const unsigned so0 = ((unsigned)b * 2048u + (unsigned)pr[0]) * 2048u;
      const unsigned so1 = ((unsigned)b * 2048u + (unsigned)pr[1]) * 2048u;
      const unsigned drow = ldsA + (unsigned)(row * (ROWH * 2));
      #pragma unroll
      for (int cc = 0; cc < 4; ++cc) {
        const unsigned co = (unsigned)((cc * 32 + lane) * 16);
        unsigned d0 = drow + co;
        const char* g0 = hsb + so0 + co;
        asm volatile("global_load_async_to_lds_b128 %0, %1, off"
                     :: "v"(d0), "v"(g0) : "memory");
        unsigned d1 = drow + 2048u + co;
        const char* g1 = hsb + so1 + co;
        asm volatile("global_load_async_to_lds_b128 %0, %1, off"
                     :: "v"(d1), "v"(g1) : "memory");
      }
    }
    asm volatile("s_wait_asynccnt 0x0" ::: "memory");
  }
  __syncthreads();

  const int ln  = lane & 15;      // N index within tile (and M index for A rows)
  const int lh  = lane >> 4;      // lane half -> K sub-offset
  const int kof = lh * 8;

  // Per-M-tile LDS base offsets, laundered through an empty asm so the
  // compiler cannot fold them into one induction register (the 0x10100 tile
  // stride exceeds the 16-bit DS immediate). Four independent DS base regs ->
  // all K-loop ds_loads use immediate offsets, no per-group address VALUs,
  // and no WMMA->VALU WAR hazard NOPs.
  const _Float16* ap[4];
  #pragma unroll
  for (int mt = 0; mt < 4; ++mt) {
    unsigned aoff = (unsigned)(((mt * 16 + ln) * ROWH + kof) * 2);
    asm volatile("" : "+v"(aoff));
    ap[mt] = (const _Float16*)(smem + aoff);
  }

  float pl[4][8][2];              // per-lane partial logits: [mtile][vgpr_j][out]
  #pragma unroll
  for (int mt = 0; mt < 4; ++mt)
    #pragma unroll
    for (int j = 0; j < 8; ++j) { pl[mt][j][0] = 0.f; pl[mt][j][1] = 0.f; }

  // Each wave owns 8 N-chunks of 16 columns; each chunk does full K then
  // fuses bias + exact GELU + the 1024->2 contraction.
  #pragma unroll 1
  for (int c = 0; c < 8; ++c) {
    const int nbase = (wave * 8 + c) * 16;
    const int n     = nbase + ln;

    v8f acc[4];
    v8f z = {};
    acc[0] = z; acc[1] = z; acc[2] = z; acc[3] = z;

    const _Float16* bcol = W1T + (size_t)n * KD + kof;

    // 4 independent K-groups per body: fresh virtual registers per group, so
    // the scheduler overlaps group g+1 loads with group g WMMAs, and there is
    // no loop-carried fragment rotation (no mov storms).
    #pragma unroll 4
    for (int k = 0; k < KD; k += 32) {
      Frag bf;
      loadB(bf, bcol + k);
      Frag af[4];
      #pragma unroll
      for (int mt = 0; mt < 4; ++mt) {
        af[mt].p.lo = *(const v8h*)(ap[mt] + k);
        af[mt].p.hi = *(const v8h*)(ap[mt] + k + 16);
      }
      #pragma unroll
      for (int mt = 0; mt < 4; ++mt)
        acc[mt] = __builtin_amdgcn_wmma_f32_16x16x32_f16(
            false, af[mt].v, false, bf.v, (short)0, acc[mt], false, false);
    }

    // Fused epilogue: h = acc + b1[n]; g = gelu_exact(h); logits += g * W2[n].
    float  b1v = b1[n];
    float2 w2v = ((const float2*)W2)[n];
    #pragma unroll
    for (int mt = 0; mt < 4; ++mt) {
      #pragma unroll
      for (int j = 0; j < 8; ++j) {
        float hval = acc[mt][j] + b1v;
        float g = 0.5f * hval * (1.0f + erff(hval * 0.70710678118654752f));
        pl[mt][j][0] += g * w2v.x;
        pl[mt][j][1] += g * w2v.y;
      }
    }
  }

  // Reduce over the 16 lanes sharing an M row (masks stay inside each half),
  // then combine across waves via LDS float atomics.
  #pragma unroll
  for (int mt = 0; mt < 4; ++mt) {
    #pragma unroll
    for (int j = 0; j < 8; ++j) {
      #pragma unroll
      for (int o = 0; o < 2; ++o) {
        float v = pl[mt][j][o];
        v += __shfl_xor(v, 1);
        v += __shfl_xor(v, 2);
        v += __shfl_xor(v, 4);
        v += __shfl_xor(v, 8);
        if (ln == 0) {
          atomicAdd(&logitsLds[(mt * 16 + j + lh * 8) * 2 + o], v);
        }
      }
    }
  }
  __syncthreads();

  if (tid < BM * 2) {
    int o = tid & 1;
    out[(size_t)gpBase * 2 + tid] = logitsLds[tid] + b2[o];
  }
}

extern "C" void kernel_launch(void* const* d_in, const int* in_sizes, int n_in,
                              void* d_out, int out_size, void* d_ws, size_t ws_size,
                              hipStream_t stream) {
  const float*     hs    = (const float*)d_in[0];
  const long long* pairs = (const long long*)d_in[1];
  const float*     W1    = (const float*)d_in[2];
  const float*     b1    = (const float*)d_in[3];
  const float*     W2    = (const float*)d_in[4];
  const float*     b2    = (const float*)d_in[5];
  float*           out   = (float*)d_out;

  _Float16* W1T   = (_Float16*)d_ws;                              // 4 MiB
  _Float16* hsF16 = (_Float16*)((char*)d_ws + ((size_t)4 << 20)); // 32 MiB

  w1_cvt_transpose<<<(KD * HD) / 256, 256, 0, stream>>>(W1, W1T);
  hs_cvt<<<(8 * 2048 * (HD / 4)) / 256, 256, 0, stream>>>(hs, hsF16);

  size_t smem = (size_t)BM * ROWH * 2   // A panel (f16)
              + (size_t)BM * 2 * 4;     // logits accumulators
  pair_mlp<<<(8 * PP) / BM, 256, smem, stream>>>(hsF16, pairs, W1T, b1, W2, b2, out);
}
